// ForwardModelTransformer_DEPRECATED_1056561954886
// MI455X (gfx1250) — compile-verified, matrix operand order checked by reading, not run on hardware
//
#include <hip/hip_runtime.h>
#include <hip/hip_bf16.h>
#include <math.h>

// Problem dims (compile-time constants from the reference)
#define BB   4
#define TT   256
#define HH   768
#define NBLK 6
#define NHD  12
#define DHD  64
#define SS   (3 * TT)     // 768 tokens per batch
#define SDIM 256
#define ODIM 256
#define H4   (4 * HH)     // 3072

typedef __attribute__((ext_vector_type(16))) __bf16 v16bf;
typedef __attribute__((ext_vector_type(8)))  __bf16 v8bf;
typedef __attribute__((ext_vector_type(8)))  float  v8f;

__device__ __forceinline__ __bf16 f2bf(float f) {
  unsigned u = __builtin_bit_cast(unsigned, f);
  unsigned r = u + 0x7FFFu + ((u >> 16) & 1u);   // round-to-nearest-even
  unsigned short s = (unsigned short)(r >> 16);
  return __builtin_bit_cast(__bf16, s);
}

// Low 32 bits of a flat pointer into LDS = LDS byte address (ISA: LDS aperture
// truncates flat addr to addr[31:0]).
__device__ __forceinline__ unsigned lds_addr(const void* p) {
  return (unsigned)(unsigned long long)(uintptr_t)p;
}

// CDNA5 async DMA: copy 16 bytes per lane global -> LDS, tracked by ASYNCcnt.
__device__ __forceinline__ void async_copy16(unsigned lds, const void* g) {
  unsigned long long ga = (unsigned long long)(uintptr_t)g;
  asm volatile("global_load_async_to_lds_b128 %0, %1, off"
               :: "v"(lds), "v"(ga) : "memory");
}

// ---------------------------------------------------------------------------
// One-time conversion kernels (weights cross HBM once as fp32, then stay bf16
// and are fully L2-resident for all 6 layers).
// ---------------------------------------------------------------------------
__global__ __launch_bounds__(256)
void cvt_plain(const float* __restrict__ src, __bf16* __restrict__ dst, long long n)
{
  for (long long i = (long long)blockIdx.x * 256 + threadIdx.x; i < n;
       i += (long long)gridDim.x * 256)
    dst[i] = f2bf(src[i]);
}

// dst[n*K + k] = bf16(src[k*N + n]); batched over blockIdx.z (layer index)
__global__ __launch_bounds__(256)
void cvt_transpose(const float* __restrict__ src, __bf16* __restrict__ dst, int K, int N)
{
  long long base = (long long)blockIdx.z * K * N;
  src += base; dst += base;
  long long total = (long long)K * N;
  for (long long i = (long long)blockIdx.x * 256 + threadIdx.x; i < total;
       i += (long long)gridDim.x * 256) {
    int k = (int)(i / N), n = (int)(i % N);
    dst[(long long)n * K + k] = f2bf(src[i]);
  }
}

// ---------------------------------------------------------------------------
// bf16 WMMA GEMM, "TN" layout: C = act( scale*(A @ Wt^T) + bias + resid )
//   A : (M,K) bf16, row stride lda (elements)
//   Wt: (N,K) bf16 (pre-transposed weights), row stride ldw
//   cmode: 0 = fp32 C, 1 = bf16 C, 2 = bf16 C in (B,NH,D,S) v-transpose layout
// 256 threads = 8 waves; block tile 128x128, K-step 32; each wave 64x32
// (8 WMMAs with A-frag x2 / B-frag x4 register reuse).
// Staging: double-buffered async global->LDS DMA (ASYNCcnt), software pipeline
// overlapping tile k+1 loads with tile k WMMAs.
// M, K exact multiples of 128/32; N may be ragged (clamped load, guarded store).
// ---------------------------------------------------------------------------
#define BM 128
#define BN 128
#define BK 32
#define GT 256

__global__ __launch_bounds__(GT)
void gemm_bf16_tn(const __bf16* __restrict__ A, int lda, long long sA0, long long sA1,
                  const __bf16* __restrict__ Wt, int ldw, long long sW0, long long sW1,
                  const float* __restrict__ bias,
                  const float* __restrict__ resid, int ldr,
                  void* __restrict__ Cptr, int ldc, long long sC0, long long sC1,
                  int M, int N, int K, int bz1, float scale, int act, int cmode)
{
  __shared__ __bf16 As[2][BM][BK + 8];   // ping-pong; +8 pad keeps 16B alignment
  __shared__ __bf16 Bs[2][BN][BK + 8];

  const int zo = blockIdx.z / bz1, zi = blockIdx.z % bz1;
  A  += zo * sA0 + zi * sA1;
  Wt += zo * sW0 + zi * sW1;
  const long long coff = zo * sC0 + zi * sC1;

  const int m0 = blockIdx.y * BM;
  const int n0 = blockIdx.x * BN;
  const int tid  = threadIdx.x;
  const int lane = tid & 31, wave = tid >> 5;
  const int wm = wave >> 2, wn = wave & 3;       // wave grid 2 (M) x 4 (N)
  const int lrow = lane & 15, lhi = lane >> 4;

  v8f acc[4][2];
#pragma unroll
  for (int mi = 0; mi < 4; ++mi)
#pragma unroll
    for (int ni = 0; ni < 2; ++ni)
#pragma unroll
      for (int e = 0; e < 8; ++e) acc[mi][ni][e] = 0.0f;

  // issue the 4 async 16B copies per wave that stage one (A,B) K-tile
  auto stage = [&](int sb, int kk) {
#pragma unroll
    for (int c = tid; c < 512; c += GT) {            // A: 128x32 bf16
      int r = c >> 2, col = (c & 3) * 8;
      async_copy16(lds_addr(&As[sb][r][col]),
                   &A[(long long)(m0 + r) * lda + kk + col]);
    }
#pragma unroll
    for (int c = tid; c < 512; c += GT) {            // B: 128x32 bf16
      int r = c >> 2, col = (c & 3) * 8;
      int rn = n0 + r; if (rn >= N) rn = N - 1;      // branch-free ragged-N clamp
      async_copy16(lds_addr(&Bs[sb][r][col]),
                   &Wt[(long long)rn * ldw + kk + col]);
    }
  };

  stage(0, 0);
  int buf = 0;
  for (int k0 = 0; k0 < K; k0 += BK) {
    if (k0 + BK < K) {
      stage(buf ^ 1, k0 + BK);                       // overlap next tile's DMA
      asm volatile("s_wait_asynccnt 0x4" ::: "memory");  // cur tile done (FIFO)
    } else {
      asm volatile("s_wait_asynccnt 0x0" ::: "memory");
    }
    __syncthreads();                                 // cur tile visible to all

    // ---- fragments: CDNA5 16-bit 16x32 layout
    //   lanes 0-15:  K 0-7 / 16-23 ; lanes 16-31: K 8-15 / 24-31
    v16bf afr[4];
#pragma unroll
    for (int mi = 0; mi < 4; ++mi) {
      int r = wm * 64 + mi * 16 + lrow;
      const v8bf lo = *reinterpret_cast<const v8bf*>(&As[buf][r][lhi * 8]);
      const v8bf hi = *reinterpret_cast<const v8bf*>(&As[buf][r][16 + lhi * 8]);
#pragma unroll
      for (int e = 0; e < 8; ++e) { afr[mi][e] = lo[e]; afr[mi][8 + e] = hi[e]; }
    }
    v16bf bfr[2];
#pragma unroll
    for (int ni = 0; ni < 2; ++ni) {
      int r = wn * 32 + ni * 16 + lrow;
      const v8bf lo = *reinterpret_cast<const v8bf*>(&Bs[buf][r][lhi * 8]);
      const v8bf hi = *reinterpret_cast<const v8bf*>(&Bs[buf][r][16 + lhi * 8]);
#pragma unroll
      for (int e = 0; e < 8; ++e) { bfr[ni][e] = lo[e]; bfr[ni][8 + e] = hi[e]; }
    }
#pragma unroll
    for (int mi = 0; mi < 4; ++mi)
#pragma unroll
      for (int ni = 0; ni < 2; ++ni)
        acc[mi][ni] = __builtin_amdgcn_wmma_f32_16x16x32_bf16(
            false, afr[mi], false, bfr[ni], (short)0, acc[mi][ni], false, false);
    __syncthreads();                                 // all reads done pre-overwrite
    buf ^= 1;
  }

  // ---- epilogue ----
#pragma unroll
  for (int mi = 0; mi < 4; ++mi) {
#pragma unroll
    for (int ni = 0; ni < 2; ++ni) {
      int n = n0 + wn * 32 + ni * 16 + lrow;
      if (n >= N) continue;
#pragma unroll
      for (int r = 0; r < 8; ++r) {
        int m = m0 + wm * 64 + mi * 16 + r + (lhi ? 8 : 0);
        float vv = acc[mi][ni][r] * scale;
        if (bias)  vv += bias[n];
        if (resid) vv += resid[(long long)m * ldr + n];
        if (act == 1) vv = 0.5f * vv * (1.0f + erff(vv * 0.70710678118654752f));
        if (cmode == 0) {
          ((float*)Cptr)[coff + (long long)m * ldc + n] = vv;
        } else if (cmode == 1) {
          ((__bf16*)Cptr)[coff + (long long)m * ldc + n] = f2bf(vv);
        } else {
          // v-transpose store: (b, head, d, s) ; m = b*S+s, n = head*64+d
          long long ci = ((long long)(m / SS) * NHD * DHD +
                          (long long)(n >> 6) * DHD + (n & 63)) * SS + (m % SS);
          ((__bf16*)Cptr)[ci] = f2bf(vv);
        }
      }
    }
  }
}

// ---------------------------------------------------------------------------
// Wave-per-row LayerNorm over 768 cols; dual output fp32 (+ optional bf16)
// ---------------------------------------------------------------------------
__global__ __launch_bounds__(256)
void layernorm_rows(const float* __restrict__ X, float* __restrict__ Y,
                    __bf16* __restrict__ Yb,
                    const float* __restrict__ g, const float* __restrict__ b,
                    int rows)
{
  int row = blockIdx.x * 8 + (threadIdx.x >> 5);
  if (row >= rows) return;
  int lane = threadIdx.x & 31;
  const float* x = X + (long long)row * HH;
  float v[HH / 32];
  float s = 0.f;
#pragma unroll
  for (int i = 0; i < HH / 32; ++i) { v[i] = x[lane + i * 32]; s += v[i]; }
#pragma unroll
  for (int o = 16; o > 0; o >>= 1) s += __shfl_xor(s, o, 32);
  float mean = s * (1.0f / HH);
  float s2 = 0.f;
#pragma unroll
  for (int i = 0; i < HH / 32; ++i) { float d = v[i] - mean; s2 += d * d; }
#pragma unroll
  for (int o = 16; o > 0; o >>= 1) s2 += __shfl_xor(s2, o, 32);
  float rstd = rsqrtf(s2 * (1.0f / HH) + 1e-5f);
  float* y = Y + (long long)row * HH;
  __bf16* yb = Yb ? Yb + (long long)row * HH : nullptr;
#pragma unroll
  for (int i = 0; i < HH / 32; ++i) {
    int c = lane + i * 32;
    float val = (v[i] - mean) * rstd * g[c] + b[c];
    y[c] = val;
    if (yb) yb[c] = f2bf(val);
  }
}

// ---------------------------------------------------------------------------
// Wave-per-row causal+mask softmax; reads fp32 scores, writes bf16 probs
// in-place into the low half of each fp32 row (row pitch stays S*4 bytes).
// ---------------------------------------------------------------------------
__global__ __launch_bounds__(256)
void softmax_causal_bf16(float* __restrict__ scores, const int* __restrict__ mask)
{
  int row  = blockIdx.x * 8 + (threadIdx.x >> 5);   // row in (B*NH*S)
  int lane = threadIdx.x & 31;
  int i  = row % SS;
  int bh = row / SS;
  int b  = bh / NHD;
  float*  p  = scores + (long long)row * SS;
  __bf16* po = reinterpret_cast<__bf16*>(p);        // bf16 row pitch = 2*S elems
  const int* mrow = mask + (long long)b * SS;

  float v[SS / 32];
  float mx = -3.4e38f;
#pragma unroll
  for (int t = 0; t < SS / 32; ++t) {
    int j = lane + t * 32;
    bool valid = (j <= i) && (mrow[j] != 0);
    v[t] = valid ? p[j] : -3.4e38f;
    mx = fmaxf(mx, v[t]);
  }
#pragma unroll
  for (int o = 16; o > 0; o >>= 1) mx = fmaxf(mx, __shfl_xor(mx, o, 32));
  float sum = 0.f;
#pragma unroll
  for (int t = 0; t < SS / 32; ++t) {
    int j = lane + t * 32;
    bool valid = (j <= i) && (mrow[j] != 0);
    float e = valid ? __expf(v[t] - mx) : 0.f;
    v[t] = e;
    sum += e;
  }
#pragma unroll
  for (int o = 16; o > 0; o >>= 1) sum += __shfl_xor(sum, o, 32);
  float inv = 1.0f / sum;
#pragma unroll
  for (int t = 0; t < SS / 32; ++t) po[lane + t * 32] = f2bf(v[t] * inv);
}

// ---------------------------------------------------------------------------
// Embedding elementwise: tiny dyn/act projections + time-embed add.
// Row 1 (state) of each token triple was written by the embedding GEMM.
// ---------------------------------------------------------------------------
__global__ __launch_bounds__(256)
void embed_elem(const int* __restrict__ timesteps,
                const float* __restrict__ actions, const float* __restrict__ dynamics,
                const float* __restrict__ embed_time,
                const float* __restrict__ W_act, const float* __restrict__ b_act,
                const float* __restrict__ W_dyn, const float* __restrict__ b_dyn,
                float* __restrict__ hraw)
{
  int tok = blockIdx.x;                     // b*T + t
  int ts  = timesteps[tok];
  const float* te  = embed_time + (long long)ts * HH;
  const float* act = actions  + (long long)tok * 2;
  const float* dyn = dynamics + (long long)tok * 5;
  float a0 = act[0], a1 = act[1];
  float d0 = dyn[0], d1 = dyn[1], d2 = dyn[2], d3 = dyn[3], d4 = dyn[4];
  float* r0 = hraw + (long long)(3 * tok) * HH;     // dynamics token
  float* r1 = r0 + HH;                              // state token
  float* r2 = r0 + 2 * HH;                          // action token
  for (int c = threadIdx.x; c < HH; c += 256) {
    float t = te[c];
    r0[c] = d0 * W_dyn[c] + d1 * W_dyn[HH + c] + d2 * W_dyn[2 * HH + c]
          + d3 * W_dyn[3 * HH + c] + d4 * W_dyn[4 * HH + c] + b_dyn[c] + t;
    r1[c] += t;
    r2[c] = a0 * W_act[c] + a1 * W_act[HH + c] + b_act[c] + t;
  }
}

// ---------------------------------------------------------------------------
// Output heads. out = [state (B*T*OD) | action (B*T*2) | dyn (B*T*5)]
// ---------------------------------------------------------------------------
__global__ __launch_bounds__(288)
void heads_kernel(const float* __restrict__ h,
                  const float* __restrict__ W_pdyn,   const float* __restrict__ b_pdyn,
                  const float* __restrict__ W_pstate, const float* __restrict__ b_pstate,
                  const float* __restrict__ W_pact,   const float* __restrict__ b_pact,
                  float* __restrict__ out)
{
  int tok = blockIdx.x;                                   // b*T + t
  const float* h2r = h + (long long)(3 * tok + 2) * HH;   // action token
  const float* h1r = h + (long long)(3 * tok + 1) * HH;   // state token
  int i = threadIdx.x;
  float sum = 0.f;
  if (i < ODIM) {
    for (int k = 0; k < HH; ++k) sum += h2r[k] * W_pstate[(long long)k * ODIM + i];
    out[(long long)tok * ODIM + i] = sum + b_pstate[i];
  } else if (i < ODIM + 5) {
    int j = i - ODIM;
    for (int k = 0; k < HH; ++k) sum += h2r[k] * W_pdyn[k * 5 + j];
    out[(long long)BB * TT * ODIM + (long long)BB * TT * 2 + (long long)tok * 5 + j] =
        sum + b_pdyn[j];
  } else if (i < ODIM + 7) {
    int j = i - ODIM - 5;
    for (int k = 0; k < HH; ++k) sum += h1r[k] * W_pact[k * 2 + j];
    out[(long long)BB * TT * ODIM + (long long)tok * 2 + j] = tanhf(sum + b_pact[j]);
  }
}

// ---------------------------------------------------------------------------
extern "C" void kernel_launch(void* const* d_in, const int* in_sizes, int n_in,
                              void* d_out, int out_size, void* d_ws, size_t ws_size,
                              hipStream_t stream)
{
  (void)in_sizes; (void)n_in; (void)out_size; (void)ws_size;

  const int*   timesteps = (const int*)  d_in[0];
  const float* states    = (const float*)d_in[1];
  const float* actions   = (const float*)d_in[2];
  const float* dynamics  = (const float*)d_in[3];
  const int*   mask      = (const int*)  d_in[4];
  const float* embed_time= (const float*)d_in[5];
  const float* W_state   = (const float*)d_in[6];
  const float* b_state   = (const float*)d_in[7];
  const float* W_act     = (const float*)d_in[8];
  const float* b_act     = (const float*)d_in[9];
  const float* W_dyn     = (const float*)d_in[10];
  const float* b_dyn     = (const float*)d_in[11];
  const float* ln_emb_g  = (const float*)d_in[12];
  const float* ln_emb_b  = (const float*)d_in[13];
  const float* blk_Wq = (const float*)d_in[14];
  const float* blk_Wk = (const float*)d_in[15];
  const float* blk_Wv = (const float*)d_in[16];
  const float* blk_Wo = (const float*)d_in[17];
  const float* blk_bq = (const float*)d_in[18];
  const float* blk_bk = (const float*)d_in[19];
  const float* blk_bv = (const float*)d_in[20];
  const float* blk_bo = (const float*)d_in[21];
  const float* blk_W1 = (const float*)d_in[22];
  const float* blk_b1 = (const float*)d_in[23];
  const float* blk_W2 = (const float*)d_in[24];
  const float* blk_b2 = (const float*)d_in[25];
  const float* blk_ln1g = (const float*)d_in[26];
  const float* blk_ln1b = (const float*)d_in[27];
  const float* blk_ln2g = (const float*)d_in[28];
  const float* blk_ln2b = (const float*)d_in[29];
  const float* W_pdyn   = (const float*)d_in[30];
  const float* b_pdyn   = (const float*)d_in[31];
  const float* W_pstate = (const float*)d_in[32];
  const float* b_pstate = (const float*)d_in[33];
  const float* W_pact   = (const float*)d_in[34];
  const float* b_pact   = (const float*)d_in[35];

  // ---------------- workspace layout ----------------
  const long long TOK  = (long long)BB * SS;       // 3072 rows
  const long long TOKH = TOK * HH;                 // 2,359,296
  const long long SCN  = (long long)BB * NHD * SS * SS;  // 28,311,552 fp32

  float* ws = (float*)d_ws;
  float* h   = ws;                  // TOKH fp32
  float* h2  = h  + TOKH;           // TOKH fp32
  float* sc  = h2 + TOKH;           // SCN  fp32 (probs bf16 written in-place)

  __bf16* bp = (__bf16*)(sc + SCN);
  __bf16* hb   = bp;  bp += TOKH;                   // LN output, GEMM A
  __bf16* qb   = bp;  bp += TOKH;                   // also reused as attb
  __bf16* kb   = bp;  bp += TOKH;
  __bf16* vtb  = bp;  bp += TOKH;                   // (B,NH,D,S)
  __bf16* midb = bp;  bp += TOK * (long long)H4;    // MLP mid
  __bf16* stb  = bp;  bp += (long long)BB * TT * SDIM;
  __bf16* Wst  = bp;  bp += (long long)SDIM * HH;   // transposed (768,256)
  __bf16* Wqt  = bp;  bp += (long long)NBLK * HH * HH;
  __bf16* Wkt  = bp;  bp += (long long)NBLK * HH * HH;
  __bf16* Wvt  = bp;  bp += (long long)NBLK * HH * HH;
  __bf16* Wot  = bp;  bp += (long long)NBLK * HH * HH;
  __bf16* W1t  = bp;  bp += (long long)NBLK * HH * H4;  // (4H,H) per layer
  __bf16* W2t  = bp;  bp += (long long)NBLK * H4 * HH;  // (H,4H) per layer
  __bf16* attb = qb;                                // alias: q dead after scores

  const dim3 gblk(GT);
  const long long Z = 0;

  // ---------------- one-time conversions (per call) ----------------
  cvt_plain<<<dim3(1024), 256, 0, stream>>>(states, stb, (long long)BB * TT * SDIM);
  cvt_transpose<<<dim3(768), 256, 0, stream>>>(W_state, Wst, SDIM, HH);
  cvt_transpose<<<dim3(2304, 1, NBLK), 256, 0, stream>>>(blk_Wq, Wqt, HH, HH);
  cvt_transpose<<<dim3(2304, 1, NBLK), 256, 0, stream>>>(blk_Wk, Wkt, HH, HH);
  cvt_transpose<<<dim3(2304, 1, NBLK), 256, 0, stream>>>(blk_Wv, Wvt, HH, HH);
  cvt_transpose<<<dim3(2304, 1, NBLK), 256, 0, stream>>>(blk_Wo, Wot, HH, HH);
  cvt_transpose<<<dim3(9216, 1, NBLK), 256, 0, stream>>>(blk_W1, W1t, HH, H4);
  cvt_transpose<<<dim3(9216, 1, NBLK), 256, 0, stream>>>(blk_W2, W2t, H4, HH);

  // ---------------- embedding ----------------
  // state embedding GEMM -> interleaved row j=1 of h2 (ldc = 3H)
  gemm_bf16_tn<<<dim3(HH / BN, (BB * TT) / BM, 1), gblk, 0, stream>>>(
      stb, SDIM, Z, Z,  Wst, SDIM, Z, Z,  b_state,
      nullptr, 0,  h2 + HH, 3 * HH, Z, Z,
      BB * TT, HH, SDIM, 1, 1.0f, 0, 0);
  embed_elem<<<BB * TT, 256, 0, stream>>>(timesteps, actions, dynamics, embed_time,
                                          W_act, b_act, W_dyn, b_dyn, h2);
  layernorm_rows<<<(int)(TOK / 8), 256, 0, stream>>>(h2, h, hb, ln_emb_g, ln_emb_b,
                                                     (int)TOK);

  // ---------------- transformer blocks ----------------
  for (int l = 0; l < NBLK; ++l) {
    const __bf16* Wq = Wqt + (long long)l * HH * HH;
    const __bf16* Wk = Wkt + (long long)l * HH * HH;
    const __bf16* Wv = Wvt + (long long)l * HH * HH;
    const __bf16* Wo = Wot + (long long)l * HH * HH;
    const __bf16* W1 = W1t + (long long)l * HH * H4;
    const __bf16* W2 = W2t + (long long)l * H4 * HH;
    const float* bq = blk_bq + (long long)l * HH;
    const float* bk = blk_bk + (long long)l * HH;
    const float* bv = blk_bv + (long long)l * HH;
    const float* bo = blk_bo + (long long)l * HH;
    const float* b1 = blk_b1 + (long long)l * H4;
    const float* b2 = blk_b2 + (long long)l * HH;

    // Q, K -> bf16 token-major ; V -> bf16 transposed (B,NH,D,S)
    gemm_bf16_tn<<<dim3(HH / BN, (int)(TOK / BM), 1), gblk, 0, stream>>>(
        hb, HH, Z, Z,  Wq, HH, Z, Z,  bq, nullptr, 0,
        qb, HH, Z, Z,  (int)TOK, HH, HH, 1, 1.0f, 0, 1);
    gemm_bf16_tn<<<dim3(HH / BN, (int)(TOK / BM), 1), gblk, 0, stream>>>(
        hb, HH, Z, Z,  Wk, HH, Z, Z,  bk, nullptr, 0,
        kb, HH, Z, Z,  (int)TOK, HH, HH, 1, 1.0f, 0, 1);
    gemm_bf16_tn<<<dim3(HH / BN, (int)(TOK / BM), 1), gblk, 0, stream>>>(
        hb, HH, Z, Z,  Wv, HH, Z, Z,  bv, nullptr, 0,
        vtb, HH, Z, Z,  (int)TOK, HH, HH, 1, 1.0f, 0, 2);

    // scores = (q @ k^T) / 8, batched over (B, NH) -> fp32
    gemm_bf16_tn<<<dim3(SS / BN, SS / BM, BB * NHD), gblk, 0, stream>>>(
        qb, HH, (long long)SS * HH, DHD,
        kb, HH, (long long)SS * HH, DHD,
        nullptr, nullptr, 0,
        sc, SS, (long long)NHD * SS * SS, (long long)SS * SS,
        SS, SS, DHD, NHD, 0.125f, 0, 0);

    softmax_causal_bf16<<<(BB * NHD * SS) / 8, 256, 0, stream>>>(sc, mask);

    // att = probs @ v : A = bf16 probs (pitch 2S inside fp32 rows), W = vtb
    gemm_bf16_tn<<<dim3(1, SS / BM, BB * NHD), gblk, 0, stream>>>(
        (const __bf16*)sc, 2 * SS,
        (long long)NHD * SS * 2 * SS, (long long)SS * 2 * SS,
        vtb, SS, (long long)NHD * DHD * SS, (long long)DHD * SS,
        nullptr, nullptr, 0,
        attb, HH, (long long)SS * HH, DHD,
        SS, DHD, SS, NHD, 1.0f, 0, 1);

    // h2 = h + att @ Wo + bo ; LN1 -> h, hb
    gemm_bf16_tn<<<dim3(HH / BN, (int)(TOK / BM), 1), gblk, 0, stream>>>(
        attb, HH, Z, Z,  Wo, HH, Z, Z,  bo, h, HH,
        h2, HH, Z, Z,  (int)TOK, HH, HH, 1, 1.0f, 0, 0);
    layernorm_rows<<<(int)(TOK / 8), 256, 0, stream>>>(
        h2, h, hb, blk_ln1g + (long long)l * HH, blk_ln1b + (long long)l * HH,
        (int)TOK);

    // mid = gelu(h @ W1 + b1) -> bf16
    gemm_bf16_tn<<<dim3(H4 / BN, (int)(TOK / BM), 1), gblk, 0, stream>>>(
        hb, HH, Z, Z,  W1, HH, Z, Z,  b1, nullptr, 0,
        midb, H4, Z, Z,  (int)TOK, H4, HH, 1, 1.0f, 1, 1);

    // h2 = h + mid @ W2 + b2 ; LN2 -> h, hb
    gemm_bf16_tn<<<dim3(HH / BN, (int)(TOK / BM), 1), gblk, 0, stream>>>(
        midb, H4, Z, Z,  W2, H4, Z, Z,  b2, h, HH,
        h2, HH, Z, Z,  (int)TOK, HH, H4, 1, 1.0f, 0, 0);
    layernorm_rows<<<(int)(TOK / 8), 256, 0, stream>>>(
        h2, h, hb, blk_ln2g + (long long)l * HH, blk_ln2b + (long long)l * HH,
        (int)TOK);
  }

  // ---------------- output heads ----------------
  heads_kernel<<<BB * TT, 288, 0, stream>>>(h, W_pdyn, b_pdyn, W_pstate, b_pstate,
                                            W_pact, b_pact, (float*)d_out);
}
